// SSMBlock_335007449442
// MI455X (gfx1250) — matrix operand order repaired and checked by reading, not compiled, not run on hardware
//
#include <hip/hip_runtime.h>

#define HDIM   2048
#define TDIM   16384
#define CHUNK  16
#define NCHUNK (TDIM / CHUNK)   // 1024
#define NLEV   10               // log2(NCHUNK)

typedef __attribute__((ext_vector_type(16))) __bf16        v16bf;
typedef __attribute__((ext_vector_type(8)))  float          v8f;
typedef __attribute__((ext_vector_type(8)))  unsigned short ushort8;

union BFrag {
  v16bf   v;
  ushort8 h[2];
};

__device__ __forceinline__ unsigned short f2bf(float x) {
  unsigned int u = __builtin_bit_cast(unsigned int, x);
  u += 0x7FFFu + ((u >> 16) & 1u);          // round-to-nearest-even
  return (unsigned short)(u >> 16);
}
__device__ __forceinline__ float bf2f(unsigned short s) {
  unsigned int u = ((unsigned int)s) << 16;
  return __builtin_bit_cast(float, u);
}

// A-fragment (16x32 bf16): lane group kg holds K = {8kg..8kg+7} U {16+8kg..16+8kg+7}
__device__ __forceinline__ v16bf ld_fragA(const unsigned short* __restrict__ row, int k0, int kg) {
  BFrag f;
  f.h[0] = *(const ushort8*)(row + k0 + 8 * kg);
  f.h[1] = *(const ushort8*)(row + k0 + 16 + 8 * kg);
  return f.v;
}
// B-fragment (32x16 bf16): lane group kg holds K = {8kg..8kg+7} U {16+8kg..16+8kg+7}
__device__ __forceinline__ v16bf ld_fragB(const unsigned short* __restrict__ row, int k0, int kg) {
  BFrag f;
  const unsigned short* p = row + k0 + 8 * kg;
  f.h[0] = *(const ushort8*)(p);
  f.h[1] = *(const ushort8*)(p + 16);
  return f.v;
}
__device__ __forceinline__ v16bf zfrag() {
  BFrag f;
  ushort8 z = {0, 0, 0, 0, 0, 0, 0, 0};
  f.h[0] = z; f.h[1] = z;
  return f.v;
}
__device__ __forceinline__ v8f wmma_bf16(v16bf a, v16bf b, v8f c) {
  return __builtin_amdgcn_wmma_f32_16x16x32_bf16(false, a, false, b, (short)0, c, false, false);
}

// ---------------- conversion kernels ----------------
__global__ void cvt_AT(const float* __restrict__ src,
                       unsigned short* __restrict__ dstN,
                       unsigned short* __restrict__ dstT) {
  size_t idx = (size_t)blockIdx.x * blockDim.x + threadIdx.x;
  int i = (int)(idx >> 11), j = (int)(idx & 2047);
  unsigned short v = f2bf(src[idx]);
  dstN[idx] = v;
  dstT[(size_t)j * HDIM + i] = v;
}
__global__ void cvt_T(const float* __restrict__ src, unsigned short* __restrict__ dstT) {
  size_t idx = (size_t)blockIdx.x * blockDim.x + threadIdx.x;
  int i = (int)(idx >> 11), j = (int)(idx & 2047);
  dstT[(size_t)j * HDIM + i] = f2bf(src[idx]);
}
__global__ void cvt_N(const float* __restrict__ src, unsigned short* __restrict__ dst) {
  size_t idx = (size_t)blockIdx.x * blockDim.x + threadIdx.x;
  dst[idx] = f2bf(src[idx]);
}

// ---------------- generic NT GEMM, 2x4 register tiling ----------------
// D[m][n] = sum_k A[m][k]*B[n][k], bf16 out. Each wave: 32(m) x 64(n) output.
// grid: (4, M/32), block 256 (8 waves); N fixed = 2048.
__global__ __launch_bounds__(256)
void gemm_nt_bf16(const unsigned short* __restrict__ A,
                  const unsigned short* __restrict__ B,
                  unsigned short* __restrict__ D,
                  int K, int ldD) {
  const int lane = threadIdx.x & 31;
  const int wv   = threadIdx.x >> 5;
  const int nt   = blockIdx.x * 8 + wv;   // 64-wide n block (0..31)
  const int mt   = blockIdx.y;            // 32-wide m block
  const int kg   = lane >> 4;
  const int r16  = lane & 15;
  const unsigned short* arow0 = A + (size_t)(mt * 32 + r16) * K;
  const unsigned short* arow1 = arow0 + (size_t)16 * K;
  const unsigned short* brow[4];
#pragma unroll
  for (int j = 0; j < 4; ++j) brow[j] = B + (size_t)(nt * 64 + j * 16 + r16) * K;
  v8f acc[2][4];
#pragma unroll
  for (int j = 0; j < 4; ++j) { acc[0][j] = (v8f){0,0,0,0,0,0,0,0}; acc[1][j] = acc[0][j]; }
  for (int k0 = 0; k0 < K; k0 += 32) {
    v16bf fa0 = ld_fragA(arow0, k0, kg);
    v16bf fa1 = ld_fragA(arow1, k0, kg);
#pragma unroll
    for (int j = 0; j < 4; ++j) {
      v16bf fb = ld_fragB(brow[j], k0, kg);
      acc[0][j] = wmma_bf16(fa0, fb, acc[0][j]);
      acc[1][j] = wmma_bf16(fa1, fb, acc[1][j]);
    }
  }
#pragma unroll
  for (int j = 0; j < 4; ++j) {
    int n = nt * 64 + j * 16 + r16;
#pragma unroll
    for (int r = 0; r < 8; ++r) {
      size_t m0 = (size_t)(mt * 32 + r + 8 * kg);
      D[m0 * ldD + n]        = f2bf(acc[0][j][r]);
      D[(m0 + 16) * ldD + n] = f2bf(acc[1][j][r]);
    }
  }
}

// ---------------- within-chunk local prefix, 2 chunks x 4 h-tiles ----------------
// out[(c*16+i)*H + h] = u_{c16+i}[h] + sum_{k=1..15, i>=k} (A^k u_{c16+i-k})[h]
// Both chunks of a pair share the same R_k B-fragments (main reuse win).
// grid: (4, NCHUNK/2), block 256
__global__ __launch_bounds__(256)
void conv_local(const unsigned short* __restrict__ U,
                const unsigned short* __restrict__ Rall,
                float* __restrict__ out) {
  const int lane = threadIdx.x & 31;
  const int wv   = threadIdx.x >> 5;
  const int hq   = blockIdx.x * 8 + wv;   // 0..31, 64-wide h block
  const int cp   = blockIdx.y;            // chunk pair
  const int kg   = lane >> 4;
  const int r16  = lane & 15;
  const int h0   = hq * 64;
  v8f acc[2][4];
#pragma unroll
  for (int j = 0; j < 4; ++j) { acc[0][j] = (v8f){0,0,0,0,0,0,0,0}; acc[1][j] = acc[0][j]; }
  for (int k = 1; k <= 15; ++k) {
    const unsigned short* Rk = Rall + (size_t)(k - 1) * HDIM * HDIM;
    const bool ok = (r16 >= k);
    const int sh = ok ? (r16 - k) : 0;
    const unsigned short* ar0 = U + (size_t)((cp * 2 + 0) * CHUNK + sh) * HDIM;
    const unsigned short* ar1 = U + (size_t)((cp * 2 + 1) * CHUNK + sh) * HDIM;
    if (k < 15) {  // warm L2 for the next power matrix rows (global_prefetch_b8)
      __builtin_prefetch(Rk + (size_t)HDIM * HDIM + (size_t)(h0 + r16) * HDIM, 0, 1);
    }
    for (int k0 = 0; k0 < HDIM; k0 += 32) {
      v16bf a0 = ok ? ld_fragA(ar0, k0, kg) : zfrag();
      v16bf a1 = ok ? ld_fragA(ar1, k0, kg) : zfrag();
#pragma unroll
      for (int j = 0; j < 4; ++j) {
        const unsigned short* br = Rk + (size_t)(h0 + j * 16 + r16) * HDIM;
        v16bf b = ld_fragB(br, k0, kg);
        acc[0][j] = wmma_bf16(a0, b, acc[0][j]);
        acc[1][j] = wmma_bf16(a1, b, acc[1][j]);
      }
    }
  }
#pragma unroll
  for (int cc = 0; cc < 2; ++cc) {
#pragma unroll
    for (int j = 0; j < 4; ++j) {
#pragma unroll
      for (int r = 0; r < 8; ++r) {
        int t = (cp * 2 + cc) * CHUNK + r + 8 * kg;
        size_t o = (size_t)t * HDIM + h0 + j * 16 + r16;
        out[o] = acc[cc][j][r] + bf2f(U[o]);   // k=0 identity term
      }
    }
  }
}

// ---------------- extract chunk-end states to bf16 [chunk][H] ----------------
__global__ void extract_sloc(const float* __restrict__ out, unsigned short* __restrict__ S) {
  size_t idx = (size_t)blockIdx.x * blockDim.x + threadIdx.x;  // NCHUNK*HDIM
  int c = (int)(idx >> 11), h = (int)(idx & 2047);
  S[idx] = f2bf(out[((size_t)c * CHUNK + 15) * HDIM + h]);
}

// ---------------- Hillis-Steele scan level over chunk states, 2x4 tiling ----------------
// Sout[c][h] = Sin[c][h] + (M_j @ Sin[c-stride])[h] ; B-operand = Mt_j row-major
// grid: (4, NCHUNK/32), block 256
__global__ __launch_bounds__(256)
void scan_level(const unsigned short* __restrict__ Sin,
                unsigned short* __restrict__ Sout,
                const unsigned short* __restrict__ Mt,
                int stride) {
  const int lane = threadIdx.x & 31;
  const int wv   = threadIdx.x >> 5;
  const int nt   = blockIdx.x * 8 + wv;   // 64-wide h block (0..31)
  const int ct   = blockIdx.y;            // 32-wide chunk block
  const int kg   = lane >> 4;
  const int r16  = lane & 15;
  const int c0r  = ct * 32 + r16;
  const int c1r  = c0r + 16;
  const bool ok0 = c0r >= stride;
  const bool ok1 = c1r >= stride;
  const unsigned short* ar0 = Sin + (size_t)(c0r - (ok0 ? stride : 0)) * HDIM;
  const unsigned short* ar1 = Sin + (size_t)(c1r - (ok1 ? stride : 0)) * HDIM;
  v8f acc[2][4];
#pragma unroll
  for (int j = 0; j < 4; ++j) { acc[0][j] = (v8f){0,0,0,0,0,0,0,0}; acc[1][j] = acc[0][j]; }
  for (int k0 = 0; k0 < HDIM; k0 += 32) {
    v16bf fa0 = ok0 ? ld_fragA(ar0, k0, kg) : zfrag();
    v16bf fa1 = ok1 ? ld_fragA(ar1, k0, kg) : zfrag();
#pragma unroll
    for (int j = 0; j < 4; ++j) {
      const unsigned short* br = Mt + (size_t)(nt * 64 + j * 16 + r16) * HDIM;
      v16bf fb = ld_fragB(br, k0, kg);
      acc[0][j] = wmma_bf16(fa0, fb, acc[0][j]);
      acc[1][j] = wmma_bf16(fa1, fb, acc[1][j]);
    }
  }
#pragma unroll
  for (int j = 0; j < 4; ++j) {
    int n = nt * 64 + j * 16 + r16;
#pragma unroll
    for (int r = 0; r < 8; ++r) {
      size_t cc0 = (size_t)(ct * 32 + r + 8 * kg);
      size_t o0 = cc0 * HDIM + n;
      size_t o1 = (cc0 + 16) * HDIM + n;
      Sout[o0] = f2bf(acc[0][j][r] + bf2f(Sin[o0]));
      Sout[o1] = f2bf(acc[1][j][r] + bf2f(Sin[o1]));
    }
  }
}

// ---------------- carry broadcast, 2x4 tiling ----------------
// out[(c*16+i)*H + h] += (P_{i+1} @ S'[c-1])[h] ; A-operand = S' rows, B = P_{i+1} row-major
// grid: (4, NCHUNK/32, 16), block 256
__global__ __launch_bounds__(256)
void add_carry(const unsigned short* __restrict__ S,
               const unsigned short* __restrict__ Pall,
               float* __restrict__ out) {
  const int lane = threadIdx.x & 31;
  const int wv   = threadIdx.x >> 5;
  const int nt   = blockIdx.x * 8 + wv;   // 64-wide h block (0..31)
  const int ct   = blockIdx.y;            // 32-wide chunk block
  const int i    = blockIdx.z;
  const int kg   = lane >> 4;
  const int r16  = lane & 15;
  const unsigned short* Pi = Pall + (size_t)i * HDIM * HDIM;   // P_{i+1}
  const int c0r  = ct * 32 + r16;
  const int c1r  = c0r + 16;
  const bool ok0 = c0r >= 1;
  const unsigned short* ar0 = S + (size_t)(c0r - (ok0 ? 1 : 0)) * HDIM;
  const unsigned short* ar1 = S + (size_t)(c1r - 1) * HDIM;
  v8f acc[2][4];
#pragma unroll
  for (int j = 0; j < 4; ++j) { acc[0][j] = (v8f){0,0,0,0,0,0,0,0}; acc[1][j] = acc[0][j]; }
  for (int k0 = 0; k0 < HDIM; k0 += 32) {
    v16bf fa0 = ok0 ? ld_fragA(ar0, k0, kg) : zfrag();
    v16bf fa1 = ld_fragA(ar1, k0, kg);
#pragma unroll
    for (int j = 0; j < 4; ++j) {
      const unsigned short* br = Pi + (size_t)(nt * 64 + j * 16 + r16) * HDIM;
      v16bf fb = ld_fragB(br, k0, kg);
      acc[0][j] = wmma_bf16(fa0, fb, acc[0][j]);
      acc[1][j] = wmma_bf16(fa1, fb, acc[1][j]);
    }
  }
#pragma unroll
  for (int j = 0; j < 4; ++j) {
    int n = nt * 64 + j * 16 + r16;
#pragma unroll
    for (int r = 0; r < 8; ++r) {
      size_t c0 = (size_t)(ct * 32 + r + 8 * kg);
      size_t c1 = c0 + 16;
      out[(c0 * CHUNK + i) * HDIM + n] += acc[0][j][r];
      out[(c1 * CHUNK + i) * HDIM + n] += acc[1][j][r];
    }
  }
}

extern "C" void kernel_launch(void* const* d_in, const int* in_sizes, int n_in,
                              void* d_out, int out_size, void* d_ws, size_t ws_size,
                              hipStream_t stream) {
  (void)in_sizes; (void)n_in; (void)out_size; (void)ws_size;
  const float* x_f = (const float*)d_in[0];   // (T,H)
  const float* a_f = (const float*)d_in[1];   // (H,H)
  const float* b_f = (const float*)d_in[2];   // (H,H)
  float* out = (float*)d_out;

  const size_t HHe = (size_t)HDIM * HDIM;     // elements
  char* ws = (char*)d_ws;
  unsigned short* Pall = (unsigned short*)ws;            ws += 16 * HHe * 2;  // A^1..A^16 (row-major)
  unsigned short* Rall = (unsigned short*)ws;            ws += 16 * HHe * 2;  // (A^T)^1..(A^T)^16
  unsigned short* Mb   = (unsigned short*)ws;            ws += 9 * HHe * 2;   // M_1..M_9
  unsigned short* Mtb  = (unsigned short*)ws;            ws += 9 * HHe * 2;   // Mt_1..Mt_9
  unsigned short* Bt   = (unsigned short*)ws;            ws += HHe * 2;       // B^T bf16
  unsigned short* U    = (unsigned short*)ws;            ws += (size_t)TDIM * HDIM * 2;
  unsigned short* Sc0  = (unsigned short*)ws;            ws += (size_t)NCHUNK * HDIM * 2;
  unsigned short* Sc1  = (unsigned short*)ws;
  unsigned short* Xbf  = (unsigned short*)d_out;         // temp: bf16 X lives in d_out until conv pass

  // 1) conversions: Abf -> Pall[0], At -> Rall[0], Bt, Xbf
  cvt_AT<<<(int)(HHe / 256), 256, 0, stream>>>(a_f, Pall, Rall);
  cvt_T <<<(int)(HHe / 256), 256, 0, stream>>>(b_f, Bt);
  cvt_N <<<(int)((size_t)TDIM * HDIM / 256), 256, 0, stream>>>(x_f, Xbf);

  // 2) U = X @ B  (U[t][h] = sum_k X[t][k] * Bt[h][k])
  gemm_nt_bf16<<<dim3(4, TDIM / 32), 256, 0, stream>>>(Xbf, Bt, U, HDIM, HDIM);

  // 3) power chains: P_k = P_{k-1} @ A (B-op = A^T), R_k = R_{k-1} @ A^T (B-op = A)
  for (int k = 2; k <= 16; ++k) {
    gemm_nt_bf16<<<dim3(4, HDIM / 32), 256, 0, stream>>>(
        Pall + (size_t)(k - 2) * HHe, Rall, Pall + (size_t)(k - 1) * HHe, HDIM, HDIM);
    gemm_nt_bf16<<<dim3(4, HDIM / 32), 256, 0, stream>>>(
        Rall + (size_t)(k - 2) * HHe, Pall, Rall + (size_t)(k - 1) * HHe, HDIM, HDIM);
  }

  // 4) scan operators: M_0 = A^16 (=Pall[15]), Mt_0 = (A^T)^16 (=Rall[15]); square up
  const unsigned short* M0  = Pall + 15 * HHe;
  const unsigned short* Mt0 = Rall + 15 * HHe;
  for (int j = 1; j <= 9; ++j) {
    const unsigned short* Mp  = (j == 1) ? M0  : Mb  + (size_t)(j - 2) * HHe;
    const unsigned short* Mtp = (j == 1) ? Mt0 : Mtb + (size_t)(j - 2) * HHe;
    gemm_nt_bf16<<<dim3(4, HDIM / 32), 256, 0, stream>>>(Mp, Mtp, Mb + (size_t)(j - 1) * HHe, HDIM, HDIM);
    gemm_nt_bf16<<<dim3(4, HDIM / 32), 256, 0, stream>>>(Mtp, Mp, Mtb + (size_t)(j - 1) * HHe, HDIM, HDIM);
  }

  // 5) within-chunk local prefix -> d_out (fully overwrites; frees Xbf scratch)
  conv_local<<<dim3(4, NCHUNK / 2), 256, 0, stream>>>(U, Rall, out);

  // 6) chunk-end local states -> Sc0
  extract_sloc<<<(int)((size_t)NCHUNK * HDIM / 256), 256, 0, stream>>>(out, Sc0);

  // 7) Hillis-Steele scan over 1024 chunk states (10 levels, ping-pong; ends in Sc0)
  unsigned short* sin = Sc0;
  unsigned short* sout = Sc1;
  for (int j = 0; j < NLEV; ++j) {
    const unsigned short* Mtj = (j == 0) ? Mt0 : Mtb + (size_t)(j - 1) * HHe;
    scan_level<<<dim3(4, NCHUNK / 32), 256, 0, stream>>>(sin, sout, Mtj, 1 << j);
    unsigned short* t = sin; sin = sout; sout = t;
  }

  // 8) carry broadcast: h_{c*16+i} += A^{i+1} @ s'_{c-1}
  add_carry<<<dim3(4, NCHUNK / 32, 16), 256, 0, stream>>>(sin, Pall, out);
}
